// Conv1d_33981781245957
// MI455X (gfx1250) — matrix operand order repaired and checked by reading, not compiled, not run on hardware
//
#include <hip/hip_runtime.h>
#include <hip/hip_bf16.h>

// ---------------------------------------------------------------------------
// Folded StyleGAN2 upsample_conv_1d for MI455X (gfx1250, wave32, WMMA).
//   y[n,co,2s+p] = bias[co] + sum_{ci,r} W[p][r][co][ci] * x[n,ci,s-1+r]
// with W folded from conv weight w and FIR k=[.25,.75,.75,.25]:
//   g0=k0*w0, g1=k0*w1+k1*w0, g2=k0*w2+k1*w1+k2*w0,
//   g3=k1*w2+k2*w1+k3*w0, g4=k2*w2+k3*w1, g5=k3*w2
//   phase0 taps = [g1,g3,g5], phase1 taps = [g0,g2,g4]
// GEMM: M=co(16/wave-tile), N=16 positions, K=768 (r outer, ci inner),
// bf16 hi/lo split -> 3x V_WMMA_F32_16X16X32_BF16 per phase per K-chunk.
// ---------------------------------------------------------------------------

typedef __attribute__((ext_vector_type(16))) __bf16 v16bf;
typedef __attribute__((ext_vector_type(8)))  __bf16 v8bf;
typedef __attribute__((ext_vector_type(8)))  float  v8f;
typedef __attribute__((ext_vector_type(4)))  float  v4f;
typedef __attribute__((ext_vector_type(2)))  float  v2f;

#define N_B    16
#define C_IN   256
#define C_OUT  256
#define D_IN   4096
#define D_OUT  8192
#define LDS_STRIDE 264   // 256 + 8 bf16 pad -> rotates LDS banks per pos row

__device__ __forceinline__ unsigned short f2bf(float f) {
  unsigned u = __float_as_uint(f);                       // round-to-nearest-even
  return (unsigned short)((u + 0x7fffu + ((u >> 16) & 1u)) >> 16);
}
__device__ __forceinline__ float bf2f(unsigned short h) {
  return __uint_as_float(((unsigned)h) << 16);
}
__device__ __forceinline__ v16bf cat8(v8bf a, v8bf b) {
  return __builtin_shufflevector(a, b, 0,1,2,3,4,5,6,7,8,9,10,11,12,13,14,15);
}
// A fragment (16-bit A 16x32 layout): per lane, K elems {base..base+7, base+16..base+23}
__device__ __forceinline__ v16bf ldA(const unsigned short* p) {
  v8bf lo = *reinterpret_cast<const v8bf*>(p);
  v8bf hi = *reinterpret_cast<const v8bf*>(p + 16);
  return cat8(lo, hi);
}
// B fragment (16-bit B 32x16 layout): per lane, K elems {base..base+7, base+16..base+23} at fixed column
__device__ __forceinline__ v16bf ldB(const unsigned short* p) {
  v8bf lo = *reinterpret_cast<const v8bf*>(p);
  v8bf hi = *reinterpret_cast<const v8bf*>(p + 16);
  return cat8(lo, hi);
}
__device__ __forceinline__ v8f wmma_bf16(v16bf a, v16bf b, v8f c) {
  return __builtin_amdgcn_wmma_f32_16x16x32_bf16(false, a, false, b, (short)0, c,
                                                 false, false);
}

// ---- kernel 1: fold FIR into conv weights, split to bf16 hi/lo --------------
__global__ __launch_bounds__(256) void fold_weights(
    const float* __restrict__ w, unsigned short* __restrict__ Whi,
    unsigned short* __restrict__ Wlo) {
  const int ci = threadIdx.x;
  const int co = blockIdx.x;
  const float* wp = w + ((size_t)co * C_IN + ci) * 3;
  const float w0 = wp[0], w1 = wp[1], w2 = wp[2];
  const float k0 = 0.25f, k1 = 0.75f, k2 = 0.75f, k3 = 0.25f;
  const float g0 = k0 * w0;
  const float g1 = k0 * w1 + k1 * w0;
  const float g2 = k0 * w2 + k1 * w1 + k2 * w0;
  const float g3 = k1 * w2 + k2 * w1 + k3 * w0;
  const float g4 = k2 * w2 + k3 * w1;
  const float g5 = k3 * w2;
  const float taps[6] = {g1, g3, g5,   // phase 0 (even outputs), r = 0,1,2
                         g0, g2, g4};  // phase 1 (odd outputs),  r = 0,1,2
#pragma unroll
  for (int i = 0; i < 6; ++i) {        // i = p*3 + r
    const size_t idx = ((size_t)i * C_OUT + co) * C_IN + ci;
    const unsigned short h = f2bf(taps[i]);
    Whi[idx] = h;
    Wlo[idx] = f2bf(taps[i] - bf2f(h));
  }
}

// ---- kernel 2: main polyphase conv as bf16x3 WMMA GEMM ----------------------
__global__ __launch_bounds__(256) void upconv_wmma(
    const float* __restrict__ x, const float* __restrict__ bias,
    const unsigned short* __restrict__ Whi, const unsigned short* __restrict__ Wlo,
    float* __restrict__ out) {
  // x tile transposed [pos 0..17][ci 0..255], pre-split bf16 hi/lo
  __shared__ __align__(16) unsigned short xh[18 * LDS_STRIDE];
  __shared__ __align__(16) unsigned short xl[18 * LDS_STRIDE];

  const int s0   = blockIdx.x * 16;          // 16 output-pair positions
  const int n    = blockIdx.y;
  const int wid  = threadIdx.x >> 5;         // 8 waves
  const int lane = threadIdx.x & 31;
  const int co0  = blockIdx.z * 128 + wid * 16;

  // ---- stage x[n][*][s0-1 .. s0+16] into LDS, transposed + bf16-split ----
  if (lane < 18) {
    const int sg  = s0 - 1 + lane;
    const bool ib = (sg >= 0) && (sg < D_IN);
    const int pos = lane;
    for (int ci = wid; ci < C_IN; ci += 8) {
      const float v = ib ? x[((size_t)n * C_IN + ci) * D_IN + sg] : 0.0f;
      const unsigned short h = f2bf(v);
      xh[pos * LDS_STRIDE + ci] = h;
      xl[pos * LDS_STRIDE + ci] = f2bf(v - bf2f(h));
    }
  }
  __syncthreads();

  const int hhalf = lane >> 4;               // lane half selects K sub-block
  const int col   = lane & 15;               // B/N column, also A row (co) offset
  const int abase = hhalf * 8;               // A: K base {0,8}
  const int bbase = hhalf * 8;               // B: K base {0,8}

  v8f acc0 = {};                             // phase 0 (even t)
  v8f acc1 = {};                             // phase 1 (odd t)

  for (int r = 0; r < 3; ++r) {
    const unsigned short* w0h =
        Whi + ((size_t)((0 * 3 + r) * C_OUT + co0 + col)) * C_IN + abase;
    const unsigned short* w0l =
        Wlo + ((size_t)((0 * 3 + r) * C_OUT + co0 + col)) * C_IN + abase;
    const unsigned short* w1h =
        Whi + ((size_t)((1 * 3 + r) * C_OUT + co0 + col)) * C_IN + abase;
    const unsigned short* w1l =
        Wlo + ((size_t)((1 * 3 + r) * C_OUT + co0 + col)) * C_IN + abase;
    const unsigned short* bh = &xh[(col + r) * LDS_STRIDE + bbase];
    const unsigned short* bl = &xl[(col + r) * LDS_STRIDE + bbase];

#pragma unroll 2
    for (int c0 = 0; c0 < C_IN; c0 += 32) {
      const v16bf B_h = ldB(bh + c0);
      const v16bf B_l = ldB(bl + c0);
      const v16bf A0h = ldA(w0h + c0);
      const v16bf A0l = ldA(w0l + c0);
      const v16bf A1h = ldA(w1h + c0);
      const v16bf A1l = ldA(w1l + c0);
      // bf16x3: hi*hi + hi*lo + lo*hi  (lo*lo term ~2^-32, dropped)
      acc0 = wmma_bf16(A0h, B_h, acc0);
      acc0 = wmma_bf16(A0h, B_l, acc0);
      acc0 = wmma_bf16(A0l, B_h, acc0);
      acc1 = wmma_bf16(A1h, B_h, acc1);
      acc1 = wmma_bf16(A1h, B_l, acc1);
      acc1 = wmma_bf16(A1l, B_h, acc1);
    }
  }

  // ---- epilogue: bias + interleave phases (t = 2s, 2s+1 adjacent) ----
  const int cobase = co0 + abase;            // C/D row M = v + hhalf*8
  const v4f b0 = *reinterpret_cast<const v4f*>(bias + cobase);
  const v4f b1 = *reinterpret_cast<const v4f*>(bias + cobase + 4);
  const int t0 = 2 * (s0 + col);
#pragma unroll
  for (int e = 0; e < 8; ++e) {
    const float bv = (e < 4) ? b0[e] : b1[e - 4];
    v2f o;
    o.x = acc0[e] + bv;                      // even output
    o.y = acc1[e] + bv;                      // odd output
    *reinterpret_cast<v2f*>(
        out + ((size_t)(n * C_OUT + cobase + e)) * D_OUT + t0) = o;
  }
}

extern "C" void kernel_launch(void* const* d_in, const int* in_sizes, int n_in,
                              void* d_out, int out_size, void* d_ws, size_t ws_size,
                              hipStream_t stream) {
  const float* x    = (const float*)d_in[0];   // [16,256,4096]
  const float* w    = (const float*)d_in[1];   // [256,256,3]
  const float* bias = (const float*)d_in[2];   // [256]
  float* out        = (float*)d_out;           // [16,256,8192]

  unsigned short* Whi = (unsigned short*)d_ws;           // 6*256*256 bf16
  unsigned short* Wlo = Whi + 6 * C_OUT * C_IN;          // 6*256*256 bf16
  (void)in_sizes; (void)n_in; (void)out_size; (void)ws_size;

  hipLaunchKernelGGL(fold_weights, dim3(C_OUT), dim3(C_IN), 0, stream, w, Whi, Wlo);

  dim3 grid(D_IN / 16, N_B, 2);   // s-tiles x batch x co-halves
  hipLaunchKernelGGL(upconv_wmma, grid, dim3(256), 0, stream, x, bias, Whi, Wlo, out);
}
